// Head_10393820856336
// MI455X (gfx1250) — matrix operand order repaired and checked by reading, not compiled, not run on hardware
//
#include <hip/hip_runtime.h>

// ---------------------------------------------------------------------------
// Causal single-head attention for MI455X (gfx1250, wave32, WMMA bf16)
//   x:[4,4096,1024] f32; Wk/Wq/Wv:[1024,64] f32; bk/bq/bv:[64] f32
//   out:[4,4096,64] f32
// Pipeline:
//   k0: W -> bf16 W^T [3][64][1024]         (makes GEMM B-frags contiguous)
//   k1: QKV GEMM (wmma bf16, f32 acc), q pre-scaled by 1/8, v stored ^T
//   k2: flash attention, 128 q-rows/block (16/wave), 32 keys/iter,
//       K/V tiles double-buffered in LDS via async global->LDS copies
// ---------------------------------------------------------------------------

typedef __attribute__((ext_vector_type(16))) __bf16 bf16x16;
typedef __attribute__((ext_vector_type(8)))  __bf16 bf16x8;
typedef __attribute__((ext_vector_type(8)))  float  f32x8;
typedef __attribute__((ext_vector_type(4)))  int    i32x4;

#define BB 4
#define TT 4096
#define CC 1024
#define HH 64

#if defined(__has_builtin)
#if __has_builtin(__builtin_amdgcn_global_load_async_to_lds_b128)
#define HAS_ASYNC_LDS 1
#endif
#endif

static __device__ __forceinline__ __bf16 f2bf(float f) {
  unsigned u = __builtin_bit_cast(unsigned, f);
  u += 0x7FFFu + ((u >> 16) & 1u);          // round-to-nearest-even
  unsigned short h = (unsigned short)(u >> 16);
  return __builtin_bit_cast(__bf16, h);
}

static __device__ __forceinline__ bf16x8 ld8(const __bf16* p) {
  return *reinterpret_cast<const bf16x8*>(p);
}

static __device__ __forceinline__ bf16x16 cat16(bf16x8 lo, bf16x8 hi) {
  bf16x16 r;
#pragma unroll
  for (int i = 0; i < 8; ++i) { r[i] = lo[i]; r[i + 8] = hi[i]; }
  return r;
}

// 16-byte global->LDS copy: async DMA path on CDNA5, VGPR bounce otherwise.
static __device__ __forceinline__ void cp16_g2l(const __bf16* g, __bf16* l) {
#if defined(HAS_ASYNC_LDS)
  __builtin_amdgcn_global_load_async_to_lds_b128(
      (__attribute__((address_space(1))) i32x4*)(__bf16*)g,
      (__attribute__((address_space(3))) i32x4*)l, 0, 0);
#else
  *reinterpret_cast<uint4*>(l) = *reinterpret_cast<const uint4*>(g);
#endif
}

static __device__ __forceinline__ void wait_stage_and_barrier() {
#if defined(HAS_ASYNC_LDS)
  asm volatile("s_wait_asynccnt 0" ::: "memory");
#endif
  __syncthreads();
}

// --------------------------------------------------------------------------
// Kernel 0: transpose+convert weights: wt[w][h][c] = W_w[c][h] (bf16)
// w: 0=Wk, 1=Wq, 2=Wv
// --------------------------------------------------------------------------
__global__ void wt_kernel(const float* __restrict__ Wk,
                          const float* __restrict__ Wq,
                          const float* __restrict__ Wv,
                          __bf16* __restrict__ wt) {
  int idx = blockIdx.x * 256 + threadIdx.x;          // [0, 3*64*1024)
  int w = idx / (HH * CC);
  int rem = idx - w * (HH * CC);
  int h = rem / CC;
  int c = rem - h * CC;
  const float* W = (w == 0) ? Wk : (w == 1) ? Wq : Wv;
  wt[idx] = f2bf(W[c * HH + h]);
}

// --------------------------------------------------------------------------
// Kernel 1: QKV projection. One wave computes a 16-row tile of q,k,v.
//   q (scaled by 1/8) and k stored [B*T, 64] bf16; v stored vt[B][64][T] bf16
// --------------------------------------------------------------------------
__global__ void qkv_kernel(const float* __restrict__ x,
                           const __bf16* __restrict__ wt,
                           const float* __restrict__ bk,
                           const float* __restrict__ bq,
                           const float* __restrict__ bv,
                           __bf16* __restrict__ qs,
                           __bf16* __restrict__ ks,
                           __bf16* __restrict__ vt) {
  const int lane = threadIdx.x & 31;
  const int wave = threadIdx.x >> 5;
  const int tile = blockIdx.x * 8 + wave;            // 16-row tile id
  const int rowbase = tile * 16;

  const int arow = rowbase + (lane & 15);
  const float* xrow = x + (size_t)arow * CC;
  const int hi16 = (lane >= 16);

  f32x8 acc[3][4];
#pragma unroll
  for (int w = 0; w < 3; ++w)
#pragma unroll
    for (int ht = 0; ht < 4; ++ht) acc[w][ht] = (f32x8){};

  for (int kc = 0; kc < CC; kc += 32) {
    // ---- A fragment from x (fp32 -> bf16), 16-bit A lane mapping ----
    const int p0 = kc + (hi16 ? 8 : 0);
    float4 a0 = *(const float4*)(xrow + p0);
    float4 a1 = *(const float4*)(xrow + p0 + 4);
    float4 a2 = *(const float4*)(xrow + p0 + 16);
    float4 a3 = *(const float4*)(xrow + p0 + 20);
    bf16x16 afrag;
    afrag[0] = f2bf(a0.x); afrag[1] = f2bf(a0.y); afrag[2]  = f2bf(a0.z); afrag[3]  = f2bf(a0.w);
    afrag[4] = f2bf(a1.x); afrag[5] = f2bf(a1.y); afrag[6]  = f2bf(a1.z); afrag[7]  = f2bf(a1.w);
    afrag[8] = f2bf(a2.x); afrag[9] = f2bf(a2.y); afrag[10] = f2bf(a2.z); afrag[11] = f2bf(a2.w);
    afrag[12] = f2bf(a3.x); afrag[13] = f2bf(a3.y); afrag[14] = f2bf(a3.z); afrag[15] = f2bf(a3.w);

    const int koff = kc + (hi16 ? 8 : 0);
#pragma unroll
    for (int w = 0; w < 3; ++w) {
#pragma unroll
      for (int ht = 0; ht < 4; ++ht) {
        const int h = ht * 16 + (lane & 15);
        const __bf16* bp = wt + ((size_t)(w * HH + h) * CC + koff);
        bf16x16 bfrag = cat16(ld8(bp), ld8(bp + 16));
        acc[w][ht] = __builtin_amdgcn_wmma_f32_16x16x32_bf16(
            false, afrag, false, bfrag, (short)0, acc[w][ht], false, false);
      }
    }
  }

  // ---- epilogue: bias, scale q, store bf16 ----
  const int rsel = hi16 ? 8 : 0;
#pragma unroll
  for (int ht = 0; ht < 4; ++ht) {
    const int h = ht * 16 + (lane & 15);
    const float biask = bk[h], biasq = bq[h], biasv = bv[h];
#pragma unroll
    for (int jv = 0; jv < 8; ++jv) {
      const int r = rowbase + jv + rsel;
      ks[(size_t)r * HH + h] = f2bf(acc[0][ht][jv] + biask);
      qs[(size_t)r * HH + h] = f2bf((acc[1][ht][jv] + biasq) * 0.125f);
      const int b = r >> 12, t = r & (TT - 1);
      vt[((size_t)(b * HH + h)) * TT + t] = f2bf(acc[2][ht][jv] + biasv);
    }
  }
}

// --------------------------------------------------------------------------
// Kernel 2: flash attention. Block = 8 waves = 128 q-rows of one batch.
// K/V tiles (32 keys) staged to LDS once per block, double-buffered.
// --------------------------------------------------------------------------
__global__ void attn_kernel(const __bf16* __restrict__ qs,
                            const __bf16* __restrict__ ks,
                            const __bf16* __restrict__ vt,
                            float* __restrict__ out) {
  __shared__ __align__(16) __bf16 kbuf[2][32][64];   // [buf][key][h]      8 KB
  __shared__ __align__(16) __bf16 vbuf[2][64][32];   // [buf][h][key]      8 KB
  __shared__ __align__(16) __bf16 pbuf[8][16][32];   // per-wave P bounce  8 KB

  const int tid = threadIdx.x;
  const int lane = tid & 31;
  const int wave = tid >> 5;
  const int batch = blockIdx.x >> 5;                 // 32 blocks per batch
  const int qb = (blockIdx.x & 31) * 128;            // block q-row base
  const int qt = qb + wave * 16;                     // this wave's q rows
  const int hi16 = (lane >= 16);
  const int rsel = hi16 ? 8 : 0;

  const int kend_wave = qt + 16;
  const int kend_blk = qb + 128;

  // per-thread staging coordinates (256 threads move 4KB K + 4KB V per tile)
  const int k_row = tid >> 3, k_col = (tid & 7) * 8;  // K: 32 rows x 128B
  const int v_row = tid >> 2, v_col = (tid & 3) * 8;  // V: 64 rows x 64B
  const __bf16* kg_base = ks + ((size_t)batch * TT + k_row) * HH + k_col;
  const __bf16* vg_base = vt + ((size_t)batch * HH + v_row) * TT + v_col;

  auto stage = [&](int buf, int kb) {
    cp16_g2l(kg_base + (size_t)kb * HH, &kbuf[buf][k_row][k_col]);
    cp16_g2l(vg_base + kb,              &vbuf[buf][v_row][v_col]);
  };

  // ---- load Q A-fragments (two 32-deep chunks over H=64) ----
  bf16x16 aq[2];
  {
    const __bf16* qrow = qs + ((size_t)(batch * TT + qt + (lane & 15))) * HH;
#pragma unroll
    for (int c = 0; c < 2; ++c) {
      const int p0 = c * 32 + (hi16 ? 8 : 0);
      aq[c] = cat16(ld8(qrow + p0), ld8(qrow + p0 + 16));
    }
  }

  f32x8 o[4];
#pragma unroll
  for (int ht = 0; ht < 4; ++ht) o[ht] = (f32x8){};
  float mrow[8], lrow[8];
#pragma unroll
  for (int j = 0; j < 8; ++j) { mrow[j] = -1.0e30f; lrow[j] = 0.0f; }

  int buf = 0;
  stage(0, 0);
  for (int kb = 0; kb < kend_blk; kb += 32) {
    wait_stage_and_barrier();                    // current buffer ready
    if (kb + 32 < kend_blk) stage(buf ^ 1, kb + 32);   // prefetch next tile

    if (kb < kend_wave) {                        // wave-uniform predicate
      // ---- S = Q K^T for 2 key half-tiles of 16 (from LDS) ----
      f32x8 s[2];
#pragma unroll
      for (int half = 0; half < 2; ++half) {
        const int kr = half * 16 + (lane & 15);
        f32x8 sv = (f32x8){};
#pragma unroll
        for (int c = 0; c < 2; ++c) {
          const int hoff = c * 32 + (hi16 ? 8 : 0);
          bf16x16 bfrag = cat16(ld8(&kbuf[buf][kr][hoff]),
                                ld8(&kbuf[buf][kr][hoff + 16]));
          sv = __builtin_amdgcn_wmma_f32_16x16x32_bf16(
              false, aq[c], false, bfrag, (short)0, sv, false, false);
        }
        s[half] = sv;
      }

      // ---- causal mask + online softmax ----
#pragma unroll
      for (int jv = 0; jv < 8; ++jv) {
        const int r = qt + jv + rsel;
        const int c0 = kb + (lane & 15);
        float v0 = (c0 <= r) ? s[0][jv] : -1.0e30f;
        float v1 = (c0 + 16 <= r) ? s[1][jv] : -1.0e30f;
        float mx = fmaxf(v0, v1);
#pragma unroll
        for (int off = 8; off >= 1; off >>= 1)
          mx = fmaxf(mx, __shfl_xor(mx, off, 32));
        const float mnew = fmaxf(mrow[jv], mx);
        const float p0 = __expf(v0 - mnew);
        const float p1 = __expf(v1 - mnew);
        const float alpha = __expf(mrow[jv] - mnew);
        float rs = p0 + p1;
#pragma unroll
        for (int off = 8; off >= 1; off >>= 1)
          rs += __shfl_xor(rs, off, 32);
        lrow[jv] = lrow[jv] * alpha + rs;
        mrow[jv] = mnew;
#pragma unroll
        for (int ht = 0; ht < 4; ++ht) o[ht][jv] *= alpha;
        const int rl = jv + rsel;
        pbuf[wave][rl][lane & 15]        = f2bf(p0);
        pbuf[wave][rl][16 + (lane & 15)] = f2bf(p1);
      }

      // same-wave LDS RAW: drain DScnt before re-layout read
      asm volatile("s_wait_dscnt 0" ::: "memory");

      // ---- P re-layout: C-layout -> A-fragment (16x32 bf16) ----
      bf16x16 ap;
      {
        const int rl = lane & 15;
        const int p0 = hi16 ? 8 : 0;
        ap = cat16(*reinterpret_cast<const bf16x8*>(&pbuf[wave][rl][p0]),
                   *reinterpret_cast<const bf16x8*>(&pbuf[wave][rl][p0 + 16]));
      }

      // ---- O += P V  (V^T tile from LDS) ----
      const int koff = hi16 ? 8 : 0;
#pragma unroll
      for (int ht = 0; ht < 4; ++ht) {
        const int h = ht * 16 + (lane & 15);
        bf16x16 bfrag = cat16(ld8(&vbuf[buf][h][koff]),
                              ld8(&vbuf[buf][h][koff + 16]));
        o[ht] = __builtin_amdgcn_wmma_f32_16x16x32_bf16(
            false, ap, false, bfrag, (short)0, o[ht], false, false);
      }
    }
    buf ^= 1;
  }

  // ---- normalize + write out (fp32) ----
#pragma unroll
  for (int ht = 0; ht < 4; ++ht) {
    const int h = ht * 16 + (lane & 15);
#pragma unroll
    for (int jv = 0; jv < 8; ++jv) {
      const int r = qt + jv + rsel;
      out[((size_t)(batch * TT + r)) * HH + h] = o[ht][jv] / lrow[jv];
    }
  }
}

// --------------------------------------------------------------------------
extern "C" void kernel_launch(void* const* d_in, const int* in_sizes, int n_in,
                              void* d_out, int out_size, void* d_ws, size_t ws_size,
                              hipStream_t stream) {
  (void)in_sizes; (void)n_in; (void)out_size; (void)ws_size;
  const float* x  = (const float*)d_in[0];
  const float* Wk = (const float*)d_in[1];
  const float* bk = (const float*)d_in[2];
  const float* Wq = (const float*)d_in[3];
  const float* bq = (const float*)d_in[4];
  const float* Wv = (const float*)d_in[5];
  const float* bv = (const float*)d_in[6];
  float* out = (float*)d_out;

  const size_t NQ = (size_t)BB * TT * HH;            // 1,048,576 elements
  __bf16* qs = (__bf16*)d_ws;
  __bf16* ks = qs + NQ;
  __bf16* vt = ks + NQ;
  __bf16* wt = vt + NQ;                              // 3*64*1024 elements

  wt_kernel<<<(3 * HH * CC) / 256, 256, 0, stream>>>(Wk, Wq, Wv, wt);
  qkv_kernel<<<(BB * TT / 16) / 8, 256, 0, stream>>>(x, wt, bk, bq, bv, qs, ks, vt);
  attn_kernel<<<(BB * TT / 128), 256, 0, stream>>>(qs, ks, vt, out);
}
